// PerformerBlock_49108656063202
// MI455X (gfx1250) — matrix operand order repaired and checked by reading, not compiled, not run on hardware
//
#include <hip/hip_runtime.h>
#include <math.h>

// ---------------- problem constants ----------------
#define Bb   4
#define Nn   2048
#define Dm   512
#define Hh   8
#define DHd  64
#define Ll   6
#define FFd  2048
#define Mm   266          // real feature count
#define Mp   288          // padded to multiple of 32 (WMMA K) -- pads are zero
#define INNER 512         // H*DH

typedef __bf16 v16bf __attribute__((ext_vector_type(16)));
typedef float  v8f   __attribute__((ext_vector_type(8)));

__device__ __forceinline__ __bf16 f2bf(float f) {
  unsigned u = __float_as_uint(f);
  unsigned r = (u + 0x7FFFu + ((u >> 16) & 1u)) >> 16;
  unsigned short s = (unsigned short)r;
  return __builtin_bit_cast(__bf16, s);
}
__device__ __forceinline__ float bf2f(__bf16 b) {
  unsigned short s = __builtin_bit_cast(unsigned short, b);
  return __uint_as_float(((unsigned)s) << 16);
}

// CDNA5 async global->LDS copy (16B per lane), tracked by ASYNCcnt.
// LDS offset = low 32 bits of the generic pointer (flat LDS aperture rule).
__device__ __forceinline__ void async_cp16(void* lds, const void* gaddr) {
  unsigned lo = (unsigned)(unsigned long long)lds;
  asm volatile("global_load_async_to_lds_b128 %0, %1, off"
               :: "v"(lo), "v"(gaddr) : "memory");
}
__device__ __forceinline__ void async_wait0() {
  asm volatile("s_wait_asynccnt 0" ::: "memory");
}

// =======================================================================
// Generic batched bf16 GEMM:  C[i][j] = alpha * sum_k A[i][k] * Bt[j][k]
// A: rows x K (row-major, lda), Bt: cols x K (row-major, ldb) == B^T
// Block tile 128x64x32, 256 threads (8 waves, wave32).
// Wave w owns a 32x32 sub-tile: rows (w>>1)*32, cols (w&1)*32:
//   2 A-frags x 2 B-frags -> 4 v_wmma_f32_16x16x32_bf16 per K-step.
// Double-buffered LDS fed by global_load_async_to_lds_b128 (ASYNCcnt),
// overlapping DMA for tile t+1 with WMMA on tile t.
// flags: 1=+bias[col]  2=gelu  4=+res[same layout as C]  8=*rowscale[row]
//        16=store bf16
// batch: z -> (z1=z/Z2, z2=z%Z2), element offsets offX = z1*sX1 + z2*sX2
// =======================================================================
#define LDSS 40   // LDS row stride in bf16 elems (80B; 16B-aligned chunks)

__global__ __launch_bounds__(256) void k_gemm(
    const __bf16* __restrict__ A, const __bf16* __restrict__ Bt,
    void* __restrict__ Cv,
    const float* __restrict__ bias, const float* __restrict__ res,
    const float* __restrict__ rs,
    int rows, int cols, int K, int lda, int ldb, int ldc,
    long long sA1, long long sA2, long long sB1, long long sB2,
    long long sC1, long long sC2, long long sR1, long long sR2,
    int Z2, float alpha, int flags)
{
  __shared__ alignas(16) __bf16 As[2][128 * LDSS];
  __shared__ alignas(16) __bf16 Bs[2][64 * LDSS];

  const int tid  = threadIdx.x;
  const int wave = tid >> 5;
  const int lane = tid & 31;
  const int lh   = lane & 15;
  const int kh   = lane >> 4;
  const int wr   = (wave >> 1) * 32;   // wave row offset in tile
  const int wc   = (wave & 1) * 32;    // wave col offset in tile

  const int z  = blockIdx.z;
  const int z1 = z / Z2, z2 = z % Z2;
  const long long offA = z1 * sA1 + z2 * sA2;
  const long long offB = z1 * sB1 + z2 * sB2;
  const long long offC = z1 * sC1 + z2 * sC2;
  const long long offR = z1 * sR1 + z2 * sR2;

  const int rowBase = blockIdx.y * 128;
  const int colBase = blockIdx.x * 64;

  // Issue async DMA for one 128x32 A tile + 64x32 B tile into buffer `buf`.
  // OOB lanes simply don't issue: stale LDS there only feeds C elements
  // that the epilogue never stores.
  auto load_tiles = [&](int buf, int k0) {
    #pragma unroll
    for (int it = 0; it < 2; ++it) {           // A: 512 16B chunks / 256 thr
      int c  = tid + (it << 8);
      int r  = c >> 2;
      int kc = (c & 3) << 3;
      int gr = rowBase + r;
      if (gr < rows)
        async_cp16(&As[buf][r * LDSS + kc],
                   A + offA + (long long)gr * lda + k0 + kc);
    }
    {                                          // B: 256 16B chunks / 256 thr
      int r  = tid >> 2;
      int kc = (tid & 3) << 3;
      int gc = colBase + r;
      if (gc < cols)
        async_cp16(&Bs[buf][r * LDSS + kc],
                   Bt + offB + (long long)gc * ldb + k0 + kc);
    }
  };

  v8f acc[2][2];
  #pragma unroll
  for (int i = 0; i < 2; ++i)
    #pragma unroll
    for (int j = 0; j < 2; ++j)
      #pragma unroll
      for (int e = 0; e < 8; ++e) acc[i][j][e] = 0.0f;

  load_tiles(0, 0);
  async_wait0();
  __syncthreads();

  int buf = 0;
  for (int k0 = 0; k0 < K; k0 += 32) {
    if (k0 + 32 < K) load_tiles(buf ^ 1, k0 + 32);   // DMA next tile
    if (k0 + 64 < K) {                               // L2 warm 2 tiles ahead
      int pr = rowBase + (tid >> 1);
      if (pr < rows) __builtin_prefetch(A + offA + (long long)pr * lda + k0 + 64, 0, 3);
    }

    // A fragments (16x32 wave32 layout: lane row=lh, k-half=kh)
    union { v16bf v; uint4 u[2]; } a0, a1, b0, b1;
    {
      const __bf16* Ab = &As[buf][0];
      int r0 = (wr + lh) * LDSS,  r1 = (wr + 16 + lh) * LDSS;
      a0.u[0] = *reinterpret_cast<const uint4*>(&Ab[r0 + kh * 8]);
      a0.u[1] = *reinterpret_cast<const uint4*>(&Ab[r0 + 16 + kh * 8]);
      a1.u[0] = *reinterpret_cast<const uint4*>(&Ab[r1 + kh * 8]);
      a1.u[1] = *reinterpret_cast<const uint4*>(&Ab[r1 + 16 + kh * 8]);
      // B fragments (32x16: lane col=lh, k = kh*8..+7 and 16+kh*8..+7)
      const __bf16* Bbuf = &Bs[buf][0];
      int c0 = (wc + lh) * LDSS,  c1 = (wc + 16 + lh) * LDSS;
      b0.u[0] = *reinterpret_cast<const uint4*>(&Bbuf[c0 + kh * 8]);
      b0.u[1] = *reinterpret_cast<const uint4*>(&Bbuf[c0 + 16 + kh * 8]);
      b1.u[0] = *reinterpret_cast<const uint4*>(&Bbuf[c1 + kh * 8]);
      b1.u[1] = *reinterpret_cast<const uint4*>(&Bbuf[c1 + 16 + kh * 8]);
    }
    acc[0][0] = __builtin_amdgcn_wmma_f32_16x16x32_bf16(false, a0.v, false, b0.v, (short)0, acc[0][0], false, false);
    acc[0][1] = __builtin_amdgcn_wmma_f32_16x16x32_bf16(false, a0.v, false, b1.v, (short)0, acc[0][1], false, false);
    acc[1][0] = __builtin_amdgcn_wmma_f32_16x16x32_bf16(false, a1.v, false, b0.v, (short)0, acc[1][0], false, false);
    acc[1][1] = __builtin_amdgcn_wmma_f32_16x16x32_bf16(false, a1.v, false, b1.v, (short)0, acc[1][1], false, false);

    async_wait0();       // next tile landed in LDS
    __syncthreads();     // all waves done reading `buf`
    buf ^= 1;
  }

  // epilogue: D layout m = kh*8 + j, n = lh
  #pragma unroll
  for (int ri = 0; ri < 2; ++ri) {
    #pragma unroll
    for (int ci = 0; ci < 2; ++ci) {
      #pragma unroll
      for (int j = 0; j < 8; ++j) {
        int gr = rowBase + wr + ri * 16 + kh * 8 + j;
        int gc = colBase + wc + ci * 16 + lh;
        if (gr < rows && gc < cols) {
          float v = acc[ri][ci][j] * alpha;
          if (flags & 1) v += bias[gc];
          if (flags & 2) v = 0.5f * v * (1.0f + erff(v * 0.70710678118f));
          if (flags & 8) v *= rs[offR + gr];
          long long ci2 = offC + (long long)gr * ldc + gc;
          if (flags & 4) v += res[ci2];
          if (flags & 16) ((__bf16*)Cv)[ci2] = f2bf(v);
          else            ((float*)Cv)[ci2]  = v;
        }
      }
    }
  }
}

// ---------------- elementwise / reduction kernels ----------------

__global__ void k_addpos(const float* __restrict__ xin, float* __restrict__ xout, int total) {
  int i = blockIdx.x * blockDim.x + threadIdx.x;
  if (i >= total) return;
  int d = i % Dm;
  int n = (i / Dm) % Nn;
  int j = (d < Dm / 2) ? d : d - Dm / 2;
  float f = expf(-(float)(2 * j) / (float)Dm * 9.210340371976184f); // ln(1e4)
  float a = (float)n * f;
  float pe = (d < Dm / 2) ? sinf(a) : cosf(a);
  xout[i] = xin[i] + pe;
}

__global__ void k_layernorm(const float* __restrict__ x, const float* __restrict__ w,
                            const float* __restrict__ b, void* __restrict__ out, int outBf) {
  int row = blockIdx.x;     // B*N rows
  int tid = threadIdx.x;    // 128
  const float* xr = x + (long long)row * Dm;
  float vals[4], s = 0.f, s2 = 0.f;
  #pragma unroll
  for (int i = 0; i < 4; ++i) {
    float v = xr[tid + i * 128]; vals[i] = v; s += v; s2 += v * v;
  }
  __shared__ float sb[128], sq[128];
  sb[tid] = s; sq[tid] = s2; __syncthreads();
  for (int off = 64; off > 0; off >>= 1) {
    if (tid < off) { sb[tid] += sb[tid + off]; sq[tid] += sq[tid + off]; }
    __syncthreads();
  }
  float mean = sb[0] / (float)Dm;
  float var  = sq[0] / (float)Dm - mean * mean;
  float inv  = rsqrtf(var + 1e-5f);
  #pragma unroll
  for (int i = 0; i < 4; ++i) {
    int c = tid + i * 128;
    float v = (vals[i] - mean) * inv * w[c] + b[c];
    if (outBf) ((__bf16*)out)[(long long)row * Dm + c] = f2bf(v);
    else       ((float*)out)[(long long)row * Dm + c]  = v;
  }
}

// rotary embedding per (b,n,h), 32 threads = 1 wave, one pair per thread.
// also computes diag[b,h,n] = 0.5*dn^2 * sum(rot^2)  (dn^2 = 1/8 for DH=64)
__global__ void k_rotary(const float* __restrict__ in, __bf16* __restrict__ out,
                         float* __restrict__ diag) {
  int bi = blockIdx.x;
  int h = bi % Hh;
  int n = (bi / Hh) % Nn;
  int b = bi / (Hh * Nn);
  int t = threadIdx.x; // 0..31
  long long base = ((long long)(b * Nn + n)) * INNER + h * DHd;
  float q0 = in[base + 2 * t], q1 = in[base + 2 * t + 1];
  float f = expf(-(float)(2 * t) / (float)DHd * 9.210340371976184f);
  float a = (float)n * f;
  float sn = sinf(a), cs = cosf(a);
  float r0 = q0 * cs - q1 * sn;
  float r1 = q1 * cs + q0 * sn;
  out[base + 2 * t]     = f2bf(r0);
  out[base + 2 * t + 1] = f2bf(r1);
  float ss = r0 * r0 + r1 * r1;
  for (int off = 16; off > 0; off >>= 1) ss += __shfl_down(ss, off);
  if (t == 0) diag[((long long)(b * Hh + h)) * Nn + n] = ss * 0.0625f;
}

// FAVOR+ for queries: per-row max then exp; writes qp [B,H,N,Mp] bf16 (pads 0)
__global__ void k_featq(const float* __restrict__ dd, const float* __restrict__ diag,
                        __bf16* __restrict__ qp) {
  int row = blockIdx.x;   // (b*H+h)*N + n
  int tid = threadIdx.x;  // 128
  const float* dr = dd + (long long)row * Mp;
  float mx = -1e30f;
  for (int m = tid; m < Mm; m += 128) mx = fmaxf(mx, dr[m]);
  __shared__ float sb[128];
  sb[tid] = mx; __syncthreads();
  for (int off = 64; off > 0; off >>= 1) {
    if (tid < off) sb[tid] = fmaxf(sb[tid], sb[tid + off]);
    __syncthreads();
  }
  mx = sb[0];
  float dg = diag[row];
  for (int m = tid; m < Mp; m += 128) {
    float v = (m < Mm) ? 0.06131393944f * (expf(dr[m] - dg - mx) + 1e-4f) : 0.0f;
    qp[(long long)row * Mp + m] = f2bf(v);
  }
}

__global__ void k_rowmax(const float* __restrict__ dd, float* __restrict__ tmp) {
  int row = blockIdx.x; int tid = threadIdx.x;
  const float* dr = dd + (long long)row * Mp;
  float mx = -1e30f;
  for (int m = tid; m < Mm; m += 128) mx = fmaxf(mx, dr[m]);
  __shared__ float sb[128];
  sb[tid] = mx; __syncthreads();
  for (int off = 64; off > 0; off >>= 1) {
    if (tid < off) sb[tid] = fmaxf(sb[tid], sb[tid + off]);
    __syncthreads();
  }
  if (tid == 0) tmp[row] = sb[0];
}

__global__ void k_kmax(const float* __restrict__ tmp, float* __restrict__ kmax) {
  int bh = blockIdx.x; int tid = threadIdx.x; // 256
  float mx = -1e30f;
  for (int n = tid; n < Nn; n += 256) mx = fmaxf(mx, tmp[(long long)bh * Nn + n]);
  __shared__ float sb[256];
  sb[tid] = mx; __syncthreads();
  for (int off = 128; off > 0; off >>= 1) {
    if (tid < off) sb[tid] = fmaxf(sb[tid], sb[tid + off]);
    __syncthreads();
  }
  if (tid == 0) kmax[bh] = sb[0];
}

// FAVOR+ for keys: writes kp TRANSPOSED  kpt[b,h,m,n] bf16 (pad rows zero)
__global__ void k_featk(const float* __restrict__ dd, const float* __restrict__ diag,
                        const float* __restrict__ kmax, __bf16* __restrict__ kpt) {
  int row = blockIdx.x;   // bh*N + n
  int tid = threadIdx.x;  // 128
  int bh = row / Nn;
  int n  = row % Nn;
  float mx = kmax[bh];
  float dg = diag[row];
  const float* dr = dd + (long long)row * Mp;
  for (int m = tid; m < Mp; m += 128) {
    float v = (m < Mm) ? 0.06131393944f * (expf(dr[m] - dg - mx) + 1e-4f) : 0.0f;
    kpt[((long long)bh * Mp + m) * Nn + n] = f2bf(v);
  }
}

// vT[b,h,d,n] = v[b,n,h*DH+d]  (bf16)
__global__ void k_vtrans(const float* __restrict__ vf, __bf16* __restrict__ vT, int total) {
  int i = blockIdx.x * blockDim.x + threadIdx.x;
  if (i >= total) return;
  int n = i % Nn;
  int d = (i / Nn) % DHd;
  int h = (i / (Nn * DHd)) % Hh;
  int b = i / (Nn * DHd * Hh);
  vT[i] = f2bf(vf[((long long)(b * Nn + n)) * INNER + h * DHd + d]);
}

__global__ void k_ksum(const __bf16* __restrict__ kpt, float* __restrict__ ksum) {
  int rm = blockIdx.x;    // bh*Mp + m
  int tid = threadIdx.x;  // 256
  const __bf16* kr = kpt + (long long)rm * Nn;
  float s = 0.f;
  for (int n = tid; n < Nn; n += 256) s += bf2f(kr[n]);
  __shared__ float sb[256];
  sb[tid] = s; __syncthreads();
  for (int off = 128; off > 0; off >>= 1) {
    if (tid < off) sb[tid] += sb[tid + off];
    __syncthreads();
  }
  if (tid == 0) ksum[rm] = sb[0];
}

__global__ void k_dinv(const __bf16* __restrict__ qp, const float* __restrict__ ksum,
                       float* __restrict__ dinv) {
  int row = blockIdx.x;   // bh*N + n
  int tid = threadIdx.x;  // 128
  int bh = row / Nn;
  const __bf16* qr = qp + (long long)row * Mp;
  const float*  ks = ksum + (long long)bh * Mp;
  float s = 0.f;
  for (int m = tid; m < Mp; m += 128) s += bf2f(qr[m]) * ks[m];
  __shared__ float sb[128];
  sb[tid] = s; __syncthreads();
  for (int off = 64; off > 0; off >>= 1) {
    if (tid < off) sb[tid] += sb[tid + off];
    __syncthreads();
  }
  if (tid == 0) dinv[row] = 1.0f / sb[0];
}

__global__ void k_cvt(const float* __restrict__ in, __bf16* __restrict__ out, int total) {
  int i = blockIdx.x * blockDim.x + threadIdx.x;
  if (i < total) out[i] = f2bf(in[i]);
}

// batched transpose+convert: in fp32 [z][R][C] -> out bf16 [z][C][R]
__global__ void k_tconv(const float* __restrict__ in, __bf16* __restrict__ out, int R, int C) {
  long long base = (long long)blockIdx.z * R * C;
  int i = blockIdx.x * blockDim.x + threadIdx.x;
  if (i >= R * C) return;
  int r = i / C, c = i % C;
  out[base + (long long)c * R + r] = f2bf(in[base + i]);
}

// proj [L][266][64] fp32 -> projb [L][288][64] bf16 (pad rows zero)
__global__ void k_projconv(const float* __restrict__ proj, __bf16* __restrict__ out, int total) {
  int i = blockIdx.x * blockDim.x + threadIdx.x;
  if (i >= total) return;
  int d = i % DHd;
  int m = (i / DHd) % Mp;
  int l = i / (DHd * Mp);
  out[i] = (m < Mm) ? f2bf(proj[((long long)l * Mm + m) * DHd + d]) : f2bf(0.0f);
}

// ---------------- host side ----------------

static inline void launch_gemm(hipStream_t s,
    const __bf16* A, const __bf16* Bt, void* C,
    const float* bias, const float* res, const float* rs,
    int rows, int cols, int K, int lda, int ldb, int ldc,
    long long sA1, long long sA2, long long sB1, long long sB2,
    long long sC1, long long sC2, long long sR1, long long sR2,
    int Z1, int Z2, float alpha, int flags)
{
  dim3 g((cols + 63) / 64, (rows + 127) / 128, Z1 * Z2);
  k_gemm<<<g, 256, 0, s>>>(A, Bt, C, bias, res, rs, rows, cols, K, lda, ldb, ldc,
                           sA1, sA2, sB1, sB2, sC1, sC2, sR1, sR2, Z2, alpha, flags);
}

extern "C" void kernel_launch(void* const* d_in, const int* in_sizes, int n_in,
                              void* d_out, int out_size, void* d_ws, size_t ws_size,
                              hipStream_t stream) {
  (void)in_sizes; (void)n_in; (void)out_size; (void)ws_size;

  const float* x_in = (const float*)d_in[0];
  // d_in[1] = heads (compile-time constant here)
  const float* ln1w = (const float*)d_in[2];
  const float* ln1b = (const float*)d_in[3];
  const float* wq   = (const float*)d_in[4];
  const float* wk   = (const float*)d_in[5];
  const float* wv   = (const float*)d_in[6];
  const float* wo   = (const float*)d_in[7];
  const float* wob  = (const float*)d_in[8];
  const float* ln2w = (const float*)d_in[9];
  const float* ln2b = (const float*)d_in[10];
  const float* ffw1 = (const float*)d_in[11];
  const float* ffb1 = (const float*)d_in[12];
  const float* ffw2 = (const float*)d_in[13];
  const float* ffb2 = (const float*)d_in[14];
  const float* nw   = (const float*)d_in[15];
  const float* nb   = (const float*)d_in[16];
  const float* proj = (const float*)d_in[17];

  const long long BN  = (long long)Bb * Nn;            // 8192
  const long long BND = BN * Dm;                       // 4 Mi elems
  const long long BNI = BN * INNER;
  const long long BHN = (long long)Bb * Hh * Nn;       // 65536

  // ---- workspace bump allocator ----
  char* p = (char*)d_ws;
  auto alloc = [&](size_t bytes) -> void* {
    void* r = (void*)p;
    p += (bytes + 255) & ~(size_t)255;
    return r;
  };
  float*  xf    = (float*)alloc(BND * 4);
  __bf16* hb    = (__bf16*)alloc(BND * 2);
  float*  qf    = (float*)alloc(BNI * 4);
  float*  kf    = (float*)alloc(BNI * 4);
  float*  vf    = (float*)alloc(BNI * 4);
  __bf16* qrb   = (__bf16*)alloc(BNI * 2);
  __bf16* krb   = (__bf16*)alloc(BNI * 2);
  float*  diagq = (float*)alloc(BHN * 4);
  float*  diagk = (float*)alloc(BHN * 4);
  float*  dd    = (float*)alloc(BHN * Mp * 4);   // reused for q then k, then as FFN hidden
  __bf16* qp    = (__bf16*)alloc(BHN * Mp * 2);
  __bf16* kpt   = (__bf16*)alloc((long long)Bb * Hh * Mp * Nn * 2);
  float*  tmpmx = (float*)alloc(BHN * 4);
  float*  kmax  = (float*)alloc((long long)Bb * Hh * 4);
  __bf16* vT    = (__bf16*)alloc((long long)Bb * Hh * DHd * Nn * 2);
  float*  ctxT  = (float*)alloc((long long)Bb * Hh * DHd * Mp * 4);
  __bf16* ctxTb = (__bf16*)alloc((long long)Bb * Hh * DHd * Mp * 2);
  float*  ksum  = (float*)alloc((long long)Bb * Hh * Mp * 4);
  float*  dinv  = (float*)alloc(BHN * 4);
  __bf16* wqT   = (__bf16*)alloc((long long)Ll * INNER * Dm * 2);
  __bf16* wkT   = (__bf16*)alloc((long long)Ll * INNER * Dm * 2);
  __bf16* wvT   = (__bf16*)alloc((long long)Ll * INNER * Dm * 2);
  __bf16* woT   = (__bf16*)alloc((long long)Ll * Dm * INNER * 2);
  __bf16* ff1T  = (__bf16*)alloc((long long)Ll * FFd * Dm * 2);
  __bf16* ff2T  = (__bf16*)alloc((long long)Ll * Dm * FFd * 2);
  __bf16* projb = (__bf16*)alloc((long long)Ll * Mp * DHd * 2);
  // aliases (lifetimes verified disjoint within a layer)
  float*  o_f   = qf;                // attention output fp32, qf dead after rotary
  __bf16* ob    = krb;               // krb dead after dd_k GEMM
  __bf16* ffhb  = (__bf16*)dd;       // dd dead after featk; FFN hidden bf16

  // ---- weight prep (deterministic each call) ----
  {
    dim3 b(256);
    k_tconv<<<dim3(( Dm*INNER + 255)/256, 1, Ll), b, 0, stream>>>(wq,   wqT,  Dm,    INNER);
    k_tconv<<<dim3(( Dm*INNER + 255)/256, 1, Ll), b, 0, stream>>>(wk,   wkT,  Dm,    INNER);
    k_tconv<<<dim3(( Dm*INNER + 255)/256, 1, Ll), b, 0, stream>>>(wv,   wvT,  Dm,    INNER);
    k_tconv<<<dim3((INNER*Dm  + 255)/256, 1, Ll), b, 0, stream>>>(wo,   woT,  INNER, Dm);
    k_tconv<<<dim3(( Dm*FFd   + 255)/256, 1, Ll), b, 0, stream>>>(ffw1, ff1T, Dm,    FFd);
    k_tconv<<<dim3(( FFd*Dm   + 255)/256, 1, Ll), b, 0, stream>>>(ffw2, ff2T, FFd,   Dm);
    int pt = Ll * Mp * DHd;
    k_projconv<<<(pt + 255)/256, b, 0, stream>>>(proj, projb, pt);
  }

  // x = x + fixed positional embedding
  k_addpos<<<(int)((BND + 255)/256), 256, 0, stream>>>(x_in, xf, (int)BND);

  const float dn = 0.35355339059f;   // 64^-0.25

  for (int l = 0; l < Ll; ++l) {
    const __bf16* wqTl = wqT  + (long long)l * INNER * Dm;
    const __bf16* wkTl = wkT  + (long long)l * INNER * Dm;
    const __bf16* wvTl = wvT  + (long long)l * INNER * Dm;
    const __bf16* woTl = woT  + (long long)l * Dm * INNER;
    const __bf16* f1Tl = ff1T + (long long)l * FFd * Dm;
    const __bf16* f2Tl = ff2T + (long long)l * Dm * FFd;
    const __bf16* pjl  = projb + (long long)l * Mp * DHd;

    // LN1 -> hb (bf16)
    k_layernorm<<<(int)BN, 128, 0, stream>>>(xf, ln1w + l*Dm, ln1b + l*Dm, hb, 1);

    // q/k/v = hb @ W   [8192x512 @ 512x512]
    launch_gemm(stream, hb, wqTl, qf, nullptr, nullptr, nullptr,
                (int)BN, INNER, Dm, Dm, Dm, INNER, 0,0,0,0,0,0,0,0, 1,1, 1.0f, 0);
    launch_gemm(stream, hb, wkTl, kf, nullptr, nullptr, nullptr,
                (int)BN, INNER, Dm, Dm, Dm, INNER, 0,0,0,0,0,0,0,0, 1,1, 1.0f, 0);
    launch_gemm(stream, hb, wvTl, vf, nullptr, nullptr, nullptr,
                (int)BN, INNER, Dm, Dm, Dm, INNER, 0,0,0,0,0,0,0,0, 1,1, 1.0f, 0);

    // rotary (+ diag) for q and k
    k_rotary<<<(int)(Bb*Nn*Hh), 32, 0, stream>>>(qf, qrb, diagq);
    k_rotary<<<(int)(Bb*Nn*Hh), 32, 0, stream>>>(kf, krb, diagk);

    // dd = dn * (q_rot @ proj^T)  batched over (b,h): [2048x64 @ 64x288]
    launch_gemm(stream, qrb, pjl, dd, nullptr, nullptr, nullptr,
                Nn, Mp, DHd, INNER, DHd, Mp,
                (long long)Nn*INNER, DHd, 0, 0,
                (long long)Hh*Nn*Mp, (long long)Nn*Mp, 0, 0,
                Bb, Hh, dn, 0);
    k_featq<<<(int)BHN, 128, 0, stream>>>(dd, diagq, qp);

    // same for k, global max over (n,m)
    launch_gemm(stream, krb, pjl, dd, nullptr, nullptr, nullptr,
                Nn, Mp, DHd, INNER, DHd, Mp,
                (long long)Nn*INNER, DHd, 0, 0,
                (long long)Hh*Nn*Mp, (long long)Nn*Mp, 0, 0,
                Bb, Hh, dn, 0);
    k_rowmax<<<(int)BHN, 128, 0, stream>>>(dd, tmpmx);
    k_kmax<<<Bb*Hh, 256, 0, stream>>>(tmpmx, kmax);
    k_featk<<<(int)BHN, 128, 0, stream>>>(dd, diagk, kmax, kpt);

    // v head-transpose
    {
      int tot = Bb*Hh*DHd*Nn;
      k_vtrans<<<(tot + 255)/256, 256, 0, stream>>>(vf, vT, tot);
    }

    // context^T = v^T @ kp  batched (b,h): [64x2048 @ 2048x288], K = N
    launch_gemm(stream, vT, kpt, ctxT, nullptr, nullptr, nullptr,
                DHd, Mp, Nn, Nn, Nn, Mp,
                (long long)Hh*DHd*Nn, (long long)DHd*Nn,
                (long long)Hh*Mp*Nn, (long long)Mp*Nn,
                (long long)Hh*DHd*Mp, (long long)DHd*Mp, 0, 0,
                Bb, Hh, 1.0f, 0);

    // ksum, 1/denominator
    k_ksum<<<Bb*Hh*Mp, 256, 0, stream>>>(kpt, ksum);
    k_dinv<<<(int)BHN, 128, 0, stream>>>(qp, ksum, dinv);

    {
      int tot = Bb*Hh*DHd*Mp;
      k_cvt<<<(tot + 255)/256, 256, 0, stream>>>(ctxT, ctxTb, tot);
    }

    // o = (qp @ context) * dinv  batched (b,h): [2048x288 @ 288x64]
    launch_gemm(stream, qp, ctxTb, o_f, nullptr, nullptr, dinv,
                Nn, DHd, Mp, Mp, Mp, INNER,
                (long long)Hh*Nn*Mp, (long long)Nn*Mp,
                (long long)Hh*DHd*Mp, (long long)DHd*Mp,
                (long long)Nn*INNER, DHd,
                (long long)Hh*Nn, Nn,
                Bb, Hh, 1.0f, 8);

    k_cvt<<<(int)((BNI + 255)/256), 256, 0, stream>>>(o_f, ob, (int)BNI);

    // x = x + o @ wo + wo_b
    launch_gemm(stream, ob, woTl, xf, wob + l*Dm, xf, nullptr,
                (int)BN, Dm, INNER, INNER, INNER, Dm, 0,0,0,0,0,0,0,0, 1,1, 1.0f, 1|4);

    // LN2 -> hb
    k_layernorm<<<(int)BN, 128, 0, stream>>>(xf, ln2w + l*Dm, ln2b + l*Dm, hb, 1);

    // ffh = gelu(hb @ w1 + b1), stored bf16 directly
    launch_gemm(stream, hb, f1Tl, ffhb, ffb1 + l*FFd, nullptr, nullptr,
                (int)BN, FFd, Dm, Dm, Dm, FFd, 0,0,0,0,0,0,0,0, 1,1, 1.0f, 1|2|16);

    // x = x + ffh @ w2 + b2
    launch_gemm(stream, ffhb, f2Tl, xf, ffb2 + l*Dm, xf, nullptr,
                (int)BN, Dm, FFd, FFd, FFd, Dm, 0,0,0,0,0,0,0,0, 1,1, 1.0f, 1|4);
  }

  // final layernorm -> d_out (fp32)
  k_layernorm<<<(int)BN, 128, 0, stream>>>(xf, nw, nb, d_out, 0);
}
